// DecoderWithAttention_88295937671815
// MI455X (gfx1250) — matrix operand order repaired and checked by reading, not compiled, not run on hardware
//
#include <hip/hip_runtime.h>
#include <hip/hip_bf16.h>
#include <math.h>

// Sizes (fixed by the reference)
#define BB 32
#define SS 1024
#define HH 1024
#define EE 512
#define VV 32000

typedef __bf16 v16bf __attribute__((ext_vector_type(16)));
typedef __bf16 v8bf  __attribute__((ext_vector_type(8)));
typedef float  v8f   __attribute__((ext_vector_type(8)));
typedef float  v4f   __attribute__((ext_vector_type(4)));

__device__ __forceinline__ v8f wmma_bf16(v16bf a, v16bf b, v8f c) {
  return __builtin_amdgcn_wmma_f32_16x16x32_bf16(false, a, false, b, (short)0, c,
                                                 false, false);
}

__device__ __forceinline__ v8f zero_v8f() {
  v8f z = {0.f, 0.f, 0.f, 0.f, 0.f, 0.f, 0.f, 0.f};
  return z;
}

// A fragment (16x32 bf16, M x K) from a row-major bf16 matrix [m][ldk].
// lane L: m = m0 + L%16 ; element e -> k = kb + 16*(L>=16) + e
__device__ __forceinline__ v16bf frag_a(const __bf16* A, int ldk, int m0, int kb,
                                        int lane) {
  int m = m0 + (lane & 15);
  int off = kb + ((lane & 16) ? 16 : 0);
  const __bf16* p = A + (size_t)m * ldk + off;
  v8bf lo = *(const v8bf*)p;
  v8bf hi = *(const v8bf*)(p + 8);
  v16bf a;
#pragma unroll
  for (int i = 0; i < 8; ++i) { a[i] = lo[i]; a[i + 8] = hi[i]; }
  return a;
}

// B fragment (32x16 bf16, K x N) from row-major bf16 [n][ldk] (k contiguous).
// lane L: n = n0 + L%16 ; element e -> k = kb + 16*(L>=16) + e
// 16 contiguous bf16 -> one 32-byte vector load.
__device__ __forceinline__ v16bf frag_b_bf16(const __bf16* Bm, int ldk, int n, int kb,
                                             int lane) {
  int off = kb + ((lane & 16) ? 16 : 0);
  return *(const v16bf*)(Bm + (size_t)n * ldk + off);
}

// Same but converting f32 weights on the fly (single-pass weights).
__device__ __forceinline__ v16bf frag_b_f32(const float* W, int ldw, int n, int kb,
                                            int lane) {
  int off = kb + ((lane & 16) ? 16 : 0);
  const float* p = W + (size_t)n * ldw + off;
  v16bf b;
#pragma unroll
  for (int c = 0; c < 4; ++c) {
    v4f t = *(const v4f*)(p + 4 * c);
#pragma unroll
    for (int j = 0; j < 4; ++j) b[4 * c + j] = (__bf16)t[j];
  }
  return b;
}

__device__ __forceinline__ float sigmoidf_(float x) {
  return 1.0f / (1.0f + __expf(-x));
}

// ---------------------------------------------------------------------------
// K0: BmatT[n][k] = bf16(attn_W[n, H + k]) for the big enc-projection GEMM
__global__ void __launch_bounds__(256) build_bmat_kernel(
    const float* __restrict__ attn_W, __bf16* __restrict__ BmatT) {
  int idx = blockIdx.x * 256 + threadIdx.x;  // 1024*2048/4 = 524288 total
  int k4 = (idx & 511) * 4;
  int n = idx >> 9;
  v4f t = *(const v4f*)(attn_W + (size_t)n * (3 * HH) + HH + k4);
  __bf16* d = BmatT + (size_t)n * (2 * HH) + k4;
  d[0] = (__bf16)t[0]; d[1] = (__bf16)t[1];
  d[2] = (__bf16)t[2]; d[3] = (__bf16)t[3];
}

// K0b: bf16 copy of the old hidden state, A-operand for hproj / W_hh GEMMs
__global__ void __launch_bounds__(256) cvt_h_kernel(const float* __restrict__ hidden,
                                                    __bf16* __restrict__ hbf) {
  int t = blockIdx.x * 256 + threadIdx.x;  // 32768
  hbf[t] = (__bf16)hidden[t];
}

// ---------------------------------------------------------------------------
// K1: hproj[b][n] = attn_b[n] + sum_k h[b,k]*attn_W[n,k]   (k < H slice)
// One wave per n-tile; two M-tiles (b 0-15, 16-31).
__global__ void __launch_bounds__(32) hproj_kernel(
    const __bf16* __restrict__ hbf, const float* __restrict__ attn_W,
    const float* __restrict__ attn_b, float* __restrict__ hproj) {
  int lane = threadIdx.x;
  int n0 = blockIdx.x * 16;  // 64 blocks
  int n = n0 + (lane & 15);
  v8f acc0 = zero_v8f(), acc1 = zero_v8f();
  for (int kb = 0; kb < HH; kb += 32) {
    v16bf bm = frag_b_f32(attn_W, 3 * HH, n, kb, lane);
    v16bf a0 = frag_a(hbf, HH, 0, kb, lane);
    v16bf a1 = frag_a(hbf, HH, 16, kb, lane);
    acc0 = wmma_bf16(a0, bm, acc0);
    acc1 = wmma_bf16(a1, bm, acc1);
  }
  float bias = attn_b[n];
  int mb = (lane & 16) ? 8 : 0;
#pragma unroll
  for (int r = 0; r < 8; ++r) {
    hproj[(size_t)(mb + r) * HH + n] = acc0[r] + bias;
    hproj[(size_t)(mb + r + 16) * HH + n] = acc1[r] + bias;
  }
}

// ---------------------------------------------------------------------------
// K2: fused attention scores.
// scores[b,s] = sum_n vW[n] * tanh(hproj[b,n] + sum_k enc[b,s,k]*BmatT[n,k])
// Grid: 32 b x 16 s-chunks of 64 rows. Block: 256 threads (8 waves).
// LDS: 64-row x 2048-K bf16 A tile (padded), loaded once, full-K resident.
// Wave partition: 2 M-tiles x 16 N-tiles each -> 2 WMMAs per B fragment.
#define K2_ROWS 64
#define K2_LDK 2056  // 2048 + 8 bf16 pad -> 16B-aligned rows, bank-skewed
__global__ void __launch_bounds__(256) attn_scores_kernel(
    const float* __restrict__ enc, const __bf16* __restrict__ BmatT,
    const float* __restrict__ hproj, const float* __restrict__ vW,
    float* __restrict__ scores) {
  extern __shared__ char smem[];
  float* scbuf = (float*)smem;                 // 64 floats
  __bf16* As = (__bf16*)(smem + 256);          // 64 x K2_LDK bf16
  int blk = blockIdx.x;
  int b = blk >> 4;
  int s0 = (blk & 15) * K2_ROWS;
  int tid = threadIdx.x;
  if (tid < K2_ROWS) scbuf[tid] = 0.0f;

  // Stage A: 64 x 2048 f32 -> bf16 LDS (coalesced float4 reads)
  const float* encB = enc + ((size_t)b * SS + s0) * (2 * HH);
  for (int idx = tid; idx < K2_ROWS * (2 * HH) / 4; idx += 256) {
    int k4 = (idx & 511) * 4;
    int row = idx >> 9;
    v4f t = *(const v4f*)(encB + (size_t)row * (2 * HH) + k4);
    __bf16* d = As + row * K2_LDK + k4;
    d[0] = (__bf16)t[0]; d[1] = (__bf16)t[1];
    d[2] = (__bf16)t[2]; d[3] = (__bf16)t[3];
  }
  __syncthreads();

  int wave = tid >> 5, lane = tid & 31;
  int mpair = wave & 1;    // M-tiles {2*mpair, 2*mpair+1}
  int nq = wave >> 1;      // 0..3 : interleaved n-tile groups
  int lm = lane & 15;
  int ah = (lane & 16) ? 16 : 0;
  const __bf16* Arow0 = As + (mpair * 32 + lm) * K2_LDK + ah;
  const __bf16* Arow1 = Arow0 + 16 * K2_LDK;
  const __bf16* Bbase = BmatT + ((lane & 16) ? 16 : 0);

  float part0[8] = {0.f, 0.f, 0.f, 0.f, 0.f, 0.f, 0.f, 0.f};
  float part1[8] = {0.f, 0.f, 0.f, 0.f, 0.f, 0.f, 0.f, 0.f};
  for (int jt = 0; jt < 16; ++jt) {
    int n0 = (nq + 4 * jt) * 16;
    int n = n0 + lm;
    const __bf16* Bn = Bbase + (size_t)n * (2 * HH);
    v8f acc0 = zero_v8f(), acc1 = zero_v8f();
#pragma unroll 4
    for (int kb = 0; kb < 2 * HH; kb += 32) {
      v8bf a0l = *(const v8bf*)(Arow0 + kb);
      v8bf a0h = *(const v8bf*)(Arow0 + kb + 8);
      v8bf a1l = *(const v8bf*)(Arow1 + kb);
      v8bf a1h = *(const v8bf*)(Arow1 + kb + 8);
      v16bf bm = *(const v16bf*)(Bn + kb);
      v16bf a0, a1;
#pragma unroll
      for (int i = 0; i < 8; ++i) {
        a0[i] = a0l[i]; a0[i + 8] = a0h[i];
        a1[i] = a1l[i]; a1[i + 8] = a1h[i];
      }
      acc0 = wmma_bf16(a0, bm, acc0);
      acc1 = wmma_bf16(a1, bm, acc1);
    }
    float hp = hproj[(size_t)b * HH + n];
    float vw = vW[n];
#pragma unroll
    for (int r = 0; r < 8; ++r) {
      part0[r] += vw * tanhf(acc0[r] + hp);
      part1[r] += vw * tanhf(acc1[r] + hp);
    }
  }
  // Reduce over the 16 lanes that share the same row-set (different n)
#pragma unroll
  for (int r = 0; r < 8; ++r) {
    float v0 = part0[r], v1 = part1[r];
    for (int msk = 1; msk < 16; msk <<= 1) {
      v0 += __shfl_xor(v0, msk, 32);
      v1 += __shfl_xor(v1, msk, 32);
    }
    part0[r] = v0;
    part1[r] = v1;
  }
  if (lm == 0) {
    int mbase0 = mpair * 32 + ((lane & 16) ? 8 : 0);
#pragma unroll
    for (int r = 0; r < 8; ++r) {
      atomicAdd(&scbuf[mbase0 + r], part0[r]);
      atomicAdd(&scbuf[mbase0 + 16 + r], part1[r]);
    }
  }
  __syncthreads();
  if (tid < K2_ROWS) scores[(size_t)b * SS + s0 + tid] = scbuf[tid];
}

// ---------------------------------------------------------------------------
// K3a: softmax over S per batch row
__global__ void __launch_bounds__(256) softmax_kernel(
    const float* __restrict__ scores, float* __restrict__ weights) {
  __shared__ float red[256];
  int b = blockIdx.x, tid = threadIdx.x;
  const float* s = scores + (size_t)b * SS;
  float v[4], vmax = -3.402823466e+38f;
#pragma unroll
  for (int i = 0; i < 4; ++i) {
    v[i] = s[tid + 256 * i];
    vmax = fmaxf(vmax, v[i]);
  }
  red[tid] = vmax;
  __syncthreads();
  for (int off = 128; off; off >>= 1) {
    if (tid < off) red[tid] = fmaxf(red[tid], red[tid + off]);
    __syncthreads();
  }
  vmax = red[0];
  __syncthreads();
  float sum = 0.f;
#pragma unroll
  for (int i = 0; i < 4; ++i) {
    v[i] = __expf(v[i] - vmax);
    sum += v[i];
  }
  red[tid] = sum;
  __syncthreads();
  for (int off = 128; off; off >>= 1) {
    if (tid < off) red[tid] += red[tid + off];
    __syncthreads();
  }
  float inv = 1.0f / red[0];
#pragma unroll
  for (int i = 0; i < 4; ++i)
    weights[(size_t)b * SS + tid + 256 * i] = v[i] * inv;
}

// K3b: context[b,d] = sum_s weights[b,s]*enc[b,s,d]  (bandwidth bound)
__global__ void __launch_bounds__(256) context_kernel(
    const float* __restrict__ enc, const float* __restrict__ weights,
    float* __restrict__ context) {
  __shared__ float cbuf[256];
  int blk = blockIdx.x;  // 32 b x 16 d-chunks
  int b = blk >> 4;
  int d0 = (blk & 15) * 128;
  int tid = threadIdx.x;
  int d = d0 + (tid & 127);
  int sh = tid >> 7;  // s-half
  const float* encB = enc + (size_t)b * SS * (2 * HH);
  const float* wB = weights + (size_t)b * SS;
  float acc = 0.f;
  for (int s = sh * 512; s < sh * 512 + 512; ++s)
    acc += wB[s] * encB[(size_t)s * (2 * HH) + d];
  cbuf[tid] = acc;
  __syncthreads();
  if (tid < 128) context[(size_t)b * (2 * HH) + d0 + tid] = cbuf[tid] + cbuf[tid + 128];
}

// ---------------------------------------------------------------------------
// K4a: lstm_in[b][k] bf16 = [ emb[x[b]] (E) | context (2H) ]
__global__ void __launch_bounds__(256) lstm_in_kernel(
    const int* __restrict__ x, const float* __restrict__ emb,
    const float* __restrict__ context, __bf16* __restrict__ lstm_in) {
  int t = blockIdx.x * 256 + threadIdx.x;  // 32 * 2560
  int b = t / (EE + 2 * HH);
  int k = t - b * (EE + 2 * HH);
  float v;
  if (k < EE)
    v = emb[(size_t)x[b] * EE + k];
  else
    v = context[(size_t)b * (2 * HH) + (k - EE)];
  lstm_in[t] = (__bf16)v;
}

// K4b: gates[b][j] = b_ih[j]+b_hh[j] + lstm_in.W_ih[j,:] + h.W_hh[j,:]
// One wave per n-tile (4 waves / block), two M-tiles of b.
__global__ void __launch_bounds__(128) gates_kernel(
    const __bf16* __restrict__ lstm_in, const __bf16* __restrict__ hbf,
    const float* __restrict__ W_ih, const float* __restrict__ W_hh,
    const float* __restrict__ b_ih, const float* __restrict__ b_hh,
    float* __restrict__ gates) {
  int wave = threadIdx.x >> 5, lane = threadIdx.x & 31;
  int ntile = blockIdx.x * 4 + wave;  // 256 tiles over 4H
  int n = ntile * 16 + (lane & 15);
  const int KI = EE + 2 * HH;  // 2560
  v8f acc0 = zero_v8f(), acc1 = zero_v8f();
  for (int kb = 0; kb < KI; kb += 32) {
    v16bf bm = frag_b_f32(W_ih, KI, n, kb, lane);
    v16bf a0 = frag_a(lstm_in, KI, 0, kb, lane);
    v16bf a1 = frag_a(lstm_in, KI, 16, kb, lane);
    acc0 = wmma_bf16(a0, bm, acc0);
    acc1 = wmma_bf16(a1, bm, acc1);
  }
  for (int kb = 0; kb < HH; kb += 32) {
    v16bf bm = frag_b_f32(W_hh, HH, n, kb, lane);
    v16bf a0 = frag_a(hbf, HH, 0, kb, lane);
    v16bf a1 = frag_a(hbf, HH, 16, kb, lane);
    acc0 = wmma_bf16(a0, bm, acc0);
    acc1 = wmma_bf16(a1, bm, acc1);
  }
  float bias = b_ih[n] + b_hh[n];
  int mb = (lane & 16) ? 8 : 0;
#pragma unroll
  for (int r = 0; r < 8; ++r) {
    gates[(size_t)(mb + r) * (4 * HH) + n] = acc0[r] + bias;
    gates[(size_t)(mb + r + 16) * (4 * HH) + n] = acc1[r] + bias;
  }
}

// K5: LSTM pointwise + outputs h_new, c_new (+ bf16 h_new for fc GEMM)
__global__ void __launch_bounds__(256) lstm_pointwise_kernel(
    const float* __restrict__ gates, const float* __restrict__ cell,
    float* __restrict__ h_out, float* __restrict__ c_out,
    __bf16* __restrict__ hn_bf) {
  int t = blockIdx.x * 256 + threadIdx.x;  // 32768
  int b = t >> 10, n = t & 1023;
  const float* g = gates + (size_t)b * (4 * HH);
  float gi = g[n], gf = g[n + HH], gg = g[n + 2 * HH], go = g[n + 3 * HH];
  float cn = sigmoidf_(gf) * cell[t] + sigmoidf_(gi) * tanhf(gg);
  float hn = sigmoidf_(go) * tanhf(cn);
  c_out[t] = cn;
  h_out[t] = hn;
  hn_bf[t] = (__bf16)hn;
}

// ---------------------------------------------------------------------------
// K6: prediction[b][v] = fc_b[v] + h_new . fc_W[v,:]
__global__ void __launch_bounds__(128) fc_kernel(
    const __bf16* __restrict__ hn_bf, const float* __restrict__ fc_W,
    const float* __restrict__ fc_b, float* __restrict__ pred) {
  int wave = threadIdx.x >> 5, lane = threadIdx.x & 31;
  int ntile = blockIdx.x * 4 + wave;  // 2000 tiles over V
  int n = ntile * 16 + (lane & 15);
  v8f acc0 = zero_v8f(), acc1 = zero_v8f();
  for (int kb = 0; kb < HH; kb += 32) {
    v16bf bm = frag_b_f32(fc_W, HH, n, kb, lane);
    v16bf a0 = frag_a(hn_bf, HH, 0, kb, lane);
    v16bf a1 = frag_a(hn_bf, HH, 16, kb, lane);
    acc0 = wmma_bf16(a0, bm, acc0);
    acc1 = wmma_bf16(a1, bm, acc1);
  }
  float bias = fc_b[n];
  int mb = (lane & 16) ? 8 : 0;
#pragma unroll
  for (int r = 0; r < 8; ++r) {
    pred[(size_t)(mb + r) * VV + n] = acc0[r] + bias;
    pred[(size_t)(mb + r + 16) * VV + n] = acc1[r] + bias;
  }
}

// ---------------------------------------------------------------------------
extern "C" void kernel_launch(void* const* d_in, const int* in_sizes, int n_in,
                              void* d_out, int out_size, void* d_ws, size_t ws_size,
                              hipStream_t stream) {
  (void)in_sizes; (void)n_in; (void)out_size; (void)ws_size;
  const int* x = (const int*)d_in[0];
  const float* hidden = (const float*)d_in[1];
  const float* cell = (const float*)d_in[2];
  const float* enc = (const float*)d_in[3];
  const float* emb = (const float*)d_in[4];
  const float* attn_W = (const float*)d_in[5];
  const float* attn_b = (const float*)d_in[6];
  const float* v_W = (const float*)d_in[7];
  const float* W_ih = (const float*)d_in[8];
  const float* W_hh = (const float*)d_in[9];
  const float* b_ih = (const float*)d_in[10];
  const float* b_hh = (const float*)d_in[11];
  const float* fc_W = (const float*)d_in[12];
  const float* fc_b = (const float*)d_in[13];

  float* out = (float*)d_out;
  float* pred = out;                          // B*V
  float* h_out = out + (size_t)BB * VV;       // B*H
  float* c_out = h_out + (size_t)BB * HH;     // B*H
  float* w_out = c_out + (size_t)BB * HH;     // B*S

  char* w = (char*)d_ws;
  __bf16* BmatT = (__bf16*)w;      w += (size_t)HH * (2 * HH) * 2;   // 4 MB
  float* hproj = (float*)w;        w += (size_t)BB * HH * 4;
  float* scores = (float*)w;       w += (size_t)BB * SS * 4;
  float* context = (float*)w;      w += (size_t)BB * (2 * HH) * 4;
  __bf16* lstm_in = (__bf16*)w;    w += (size_t)BB * (EE + 2 * HH) * 2;
  __bf16* hbf = (__bf16*)w;        w += (size_t)BB * HH * 2;
  float* gates = (float*)w;        w += (size_t)BB * (4 * HH) * 4;
  __bf16* hn_bf = (__bf16*)w;      w += (size_t)BB * HH * 2;

  build_bmat_kernel<<<2048, 256, 0, stream>>>(attn_W, BmatT);
  cvt_h_kernel<<<128, 256, 0, stream>>>(hidden, hbf);
  hproj_kernel<<<64, 32, 0, stream>>>(hbf, attn_W, attn_b, hproj);

  size_t smem2 = 256 + (size_t)K2_ROWS * K2_LDK * 2;  // 263,424 B < 320 KB
  (void)hipFuncSetAttribute((const void*)attn_scores_kernel,
                            hipFuncAttributeMaxDynamicSharedMemorySize,
                            (int)smem2);
  attn_scores_kernel<<<512, 256, smem2, stream>>>(enc, BmatT, hproj, v_W, scores);

  softmax_kernel<<<32, 256, 0, stream>>>(scores, w_out);
  context_kernel<<<512, 256, 0, stream>>>(enc, w_out, context);

  lstm_in_kernel<<<320, 256, 0, stream>>>(x, emb, context, lstm_in);
  gates_kernel<<<64, 128, 0, stream>>>(lstm_in, hbf, W_ih, W_hh, b_ih, b_hh, gates);
  lstm_pointwise_kernel<<<128, 256, 0, stream>>>(gates, cell, h_out, c_out, hn_bf);

  fc_kernel<<<500, 128, 0, stream>>>(hn_bf, fc_W, fc_b, pred);
}
